// GNN_mol_20641612825180
// MI455X (gfx1250) — compile-verified, matrix operand order checked by reading, not run on hardware
//
#include <hip/hip_runtime.h>
#include <hip/hip_bf16.h>

// ---------------- problem constants ----------------
#define NN 100000        // nodes
#define EE 262144        // edges
#define CC 300           // channels
#define LL 5             // layers
#define GG 10            // groupnorm groups (30 ch each)
#define KP 320           // K padded to multiple of 32 (10 chunks of 32)
#define CP 304           // output channels padded to multiple of 16 (19 tiles)
#define NTILES 6250      // NN/16 row tiles (exact)
#define CTILES 19        // ceil(300/16)
#define KCHUNKS 10       // KP/32
#define EPSGN 1e-5f

typedef __attribute__((ext_vector_type(16))) __bf16 bf16x16;
typedef __attribute__((ext_vector_type(8)))  float  f32x8;

// ---------------- degree / norm ----------------
__global__ __launch_bounds__(256) void deg_init_kernel(float* __restrict__ deg) {
    int n = blockIdx.x * 256 + threadIdx.x;
    if (n < NN) deg[n] = 1.0f;                    // segment_sum(ones) + 1
}

__global__ __launch_bounds__(256) void deg_scatter_kernel(const int* __restrict__ row,
                                                          float* __restrict__ deg) {
    int e = blockIdx.x * 256 + threadIdx.x;
    if (e < EE) atomicAdd(&deg[row[e]], 1.0f);
}

__global__ __launch_bounds__(256) void dinv_kernel(const float* __restrict__ deg,
                                                   float* __restrict__ dinv) {
    int n = blockIdx.x * 256 + threadIdx.x;
    if (n < NN) dinv[n] = rsqrtf(deg[n]);
}

__global__ __launch_bounds__(256) void norm_kernel(const int* __restrict__ row,
                                                   const int* __restrict__ col,
                                                   const float* __restrict__ dinv,
                                                   float* __restrict__ norm) {
    int e = blockIdx.x * 256 + threadIdx.x;
    if (e < EE) norm[e] = dinv[row[e]] * dinv[col[e]];
}

// ---------------- bf16 packing ----------------
__global__ __launch_bounds__(256) void pack_x_kernel(const float* __restrict__ x,
                                                     __bf16* __restrict__ hb) {
    unsigned t = blockIdx.x * 256u + threadIdx.x;      // NN*KP = 32,000,000
    unsigned n = t / KP, k = t % KP;
    float v = (k < CC) ? x[(unsigned long long)n * CC + k] : 0.0f;
    hb[t] = (__bf16)v;
}

__global__ __launch_bounds__(256) void pack_w_kernel(const float* __restrict__ W,
                                                     __bf16* __restrict__ Wp) {
    unsigned t = blockIdx.x * 256u + threadIdx.x;      // CP*KP = 97,280
    if (t >= (unsigned)(CP * KP)) return;
    unsigned c = t / KP, k = t % KP;
    float v = (c < CC && k < CC) ? W[c * CC + k] : 0.0f;
    Wp[t] = (__bf16)v;
}

// ---------------- WMMA GEMM: hx[16 x 300] = hb(16x320) * Wp^T + bias ----------
// One wave per 16-row tile. A tile (16x320 bf16) resident in 80 VGPRs.
// B stream double-buffered: WMMA(k) overlaps the load of chunk k+1.
__global__ __launch_bounds__(32) void gemm_kernel(const __bf16* __restrict__ hb,
                                                  const __bf16* __restrict__ Wp,
                                                  const float*  __restrict__ bias,
                                                  float* __restrict__ hx) {
    const int lane = threadIdx.x;          // 0..31
    const int m16  = lane & 15;            // row / column within the 16x16 tile
    const int hi   = lane >> 4;            // K sub-block selector (ISA 16-bit striping)
    const int rowTile = blockIdx.x;        // 0..6249

    // ---- A: row = rowTile*16 + m16; lane covers K {hi*8..+7, hi*8+16..+23} per chunk
    const unsigned long long aBase = (unsigned long long)(rowTile * 16 + m16) * KP;
    bf16x16 A[KCHUNKS];
#pragma unroll
    for (int kc = 0; kc < KCHUNKS; ++kc) {
        const int off = kc * 32 + hi * 8;  // bf16 units; 16B aligned
        union { bf16x16 v; uint4 u[2]; } t;
        t.u[0] = *(const uint4*)(hb + aBase + off);
        t.u[1] = *(const uint4*)(hb + aBase + off + 16);
        A[kc] = t.v;
    }

    for (int ct = 0; ct < CTILES; ++ct) {
        const int colIdx = ct * 16 + m16;                 // 0..303 (pad rows in Wp)
        const float bv = bias[colIdx < CC ? colIdx : 0];  // clamp; padded cols never stored
        f32x8 acc;
#pragma unroll
        for (int r = 0; r < 8; ++r) acc[r] = bv;          // bias preloaded into C operand

        const unsigned long long bBase = (unsigned long long)colIdx * KP;
        union { bf16x16 v; uint4 u[2]; } bb[2];
        bb[0].u[0] = *(const uint4*)(Wp + bBase + hi * 8);        // prefetch chunk 0
        bb[0].u[1] = *(const uint4*)(Wp + bBase + hi * 8 + 16);
#pragma unroll
        for (int kc = 0; kc < KCHUNKS; ++kc) {
            if (kc + 1 < KCHUNKS) {                        // prefetch chunk k+1
                const int off = (kc + 1) * 32 + hi * 8;
                bb[(kc + 1) & 1].u[0] = *(const uint4*)(Wp + bBase + off);
                bb[(kc + 1) & 1].u[1] = *(const uint4*)(Wp + bBase + off + 16);
            }
            acc = __builtin_amdgcn_wmma_f32_16x16x32_bf16(
                      false, A[kc], false, bb[kc & 1].v, (short)0, acc, false, false);
        }
        if (colIdx < CC) {                                 // mask only the store tail
#pragma unroll
            for (int r = 0; r < 8; ++r) {
                const int m = r + hi * 8;                  // C/D layout: lanes>=16 hold M+8
                hx[(unsigned long long)(rowTile * 16 + m) * CC + colIdx] = acc[r];
            }
        }
    }
}

// ---------------- edge message + scatter-add (float4 per thread) --------------
// t -> edge e = t/75, channel quad c4 = (t%75)*4. 16B coalesced loads of hx/ea;
// scatter via global_atomic_add_f32 (distinct addresses in the common case).
__global__ __launch_bounds__(256) void edge_kernel(const float* __restrict__ hx,
                                                   const float* __restrict__ ea,
                                                   const int*   __restrict__ row,
                                                   const int*   __restrict__ col,
                                                   const float* __restrict__ norm,
                                                   float* __restrict__ agg) {
    unsigned t = blockIdx.x * 256u + threadIdx.x;          // EE*75 = 19,660,800
    if (t >= (unsigned)EE * (CC / 4)) return;
    unsigned e = t / (CC / 4);
    unsigned c = (t % (CC / 4)) * 4u;
    const int r = row[e], tgt = col[e];
    const float nm = norm[e];
    const float4 hv = *(const float4*)(hx + (unsigned long long)r * CC + c);
    const float4 ev = *(const float4*)(ea + (unsigned long long)e * CC + c);
    float* dst = agg + (unsigned long long)tgt * CC + c;
    atomicAdd(dst + 0, fmaxf(hv.x + ev.x, 0.0f) * nm);
    atomicAdd(dst + 1, fmaxf(hv.y + ev.y, 0.0f) * nm);
    atomicAdd(dst + 2, fmaxf(hv.z + ev.z, 0.0f) * nm);
    atomicAdd(dst + 3, fmaxf(hv.w + ev.w, 0.0f) * nm);
}

// ---------------- fused node update + GroupNorm + repack ----------------
// block = 320 threads = 10 wave32s; wave g handles group g (30 channels, lanes 0..29)
__global__ __launch_bounds__(320) void node_kernel(const float* __restrict__ agg,
                                                   const float* __restrict__ hx,
                                                   const float* __restrict__ root,
                                                   const float* __restrict__ deg,
                                                   const float* __restrict__ gamma,
                                                   const float* __restrict__ beta,
                                                   __bf16* __restrict__ hb,
                                                   float*  __restrict__ out,
                                                   int is_last) {
    const int n    = blockIdx.x;
    const int g    = threadIdx.x >> 5;
    const int lane = threadIdx.x & 31;
    const int c    = g * 30 + lane;
    const bool act = lane < 30;

    float val = 0.0f;
    if (act) {
        float hv = fmaxf(hx[(unsigned long long)n * CC + c] + root[c], 0.0f);
        val = agg[(unsigned long long)n * CC + c] + hv / deg[n];
    }
    // wave32 reduction over the 30 active lanes (inactive contribute 0)
    float s = val, q = val * val;
#pragma unroll
    for (int off = 16; off; off >>= 1) {
        s += __shfl_xor(s, off, 32);
        q += __shfl_xor(q, off, 32);
    }
    const float mean = s * (1.0f / 30.0f);
    const float var  = q * (1.0f / 30.0f) - mean * mean;
    const float rs   = rsqrtf(var + EPSGN);
    if (act) {
        float y = (val - mean) * rs * gamma[c] + beta[c];
        if (is_last) out[(unsigned long long)n * CC + c] = y;            // no ReLU on last
        else         hb[(unsigned long long)n * KP + c] = (__bf16)fmaxf(y, 0.0f);
    }
}

// ---------------- host orchestration ----------------
static inline size_t alignup(size_t v) { return (v + 255) & ~(size_t)255; }

extern "C" void kernel_launch(void* const* d_in, const int* in_sizes, int n_in,
                              void* d_out, int out_size, void* d_ws, size_t ws_size,
                              hipStream_t stream) {
    (void)in_sizes; (void)n_in; (void)out_size; (void)ws_size;
    const float* x        = (const float*)d_in[0];
    const int*   ei       = (const int*)  d_in[1];   // [2][E]
    const float* ea       = (const float*)d_in[2];   // [E][C]
    const float* W        = (const float*)d_in[3];   // [L][C][C]
    const float* b        = (const float*)d_in[4];   // [L][C]
    const float* root_emb = (const float*)d_in[5];   // [L][C]
    const float* gn_gamma = (const float*)d_in[6];   // [L][C]
    const float* gn_beta  = (const float*)d_in[7];   // [L][C]
    float* out = (float*)d_out;

    const int* row = ei;          // source j
    const int* col = ei + EE;     // target i

    // workspace carve-out
    char* p = (char*)d_ws;
    float*  deg  = (float*)p;              p += alignup(sizeof(float) * NN);
    float*  dinv = (float*)p;              p += alignup(sizeof(float) * NN);
    float*  nrm  = (float*)p;              p += alignup(sizeof(float) * EE);
    __bf16* Wp   = (__bf16*)p;             p += alignup(sizeof(__bf16) * CP * KP);
    __bf16* hb   = (__bf16*)p;             p += alignup(sizeof(__bf16) * (size_t)NN * KP);
    float*  hx   = (float*)p;              p += alignup(sizeof(float) * (size_t)NN * CC);
    float*  agg  = (float*)p;              p += alignup(sizeof(float) * (size_t)NN * CC);

    const int nb_n  = (NN + 255) / 256;
    const int nb_e  = (EE + 255) / 256;
    const int nb_px = (NN * KP) / 256;                       // exact: 125000
    const int nb_pw = (CP * KP + 255) / 256;
    const unsigned eq = (unsigned)EE * (CC / 4);             // 19,660,800 quad-threads
    const int nb_ec = (int)((eq + 255u) / 256u);

    // degree / normalization coefficients (layer-invariant)
    deg_init_kernel   <<<nb_n, 256, 0, stream>>>(deg);
    deg_scatter_kernel<<<nb_e, 256, 0, stream>>>(row, deg);
    dinv_kernel       <<<nb_n, 256, 0, stream>>>(deg, dinv);
    norm_kernel       <<<nb_e, 256, 0, stream>>>(row, col, dinv, nrm);

    // h0 = x packed to bf16
    pack_x_kernel<<<nb_px, 256, 0, stream>>>(x, hb);

    for (int l = 0; l < LL; ++l) {
        pack_w_kernel<<<nb_pw, 256, 0, stream>>>(W + (size_t)l * CC * CC, Wp);
        gemm_kernel  <<<NTILES, 32, 0, stream>>>(hb, Wp, b + (size_t)l * CC, hx);
        hipMemsetAsync(agg, 0, sizeof(float) * (size_t)NN * CC, stream);
        edge_kernel  <<<nb_ec, 256, 0, stream>>>(hx, ea, row, col, nrm, agg);
        node_kernel  <<<NN, 320, 0, stream>>>(agg, hx,
                                              root_emb + (size_t)l * CC, deg,
                                              gn_gamma + (size_t)l * CC,
                                              gn_beta  + (size_t)l * CC,
                                              hb, out, (l == LL - 1) ? 1 : 0);
    }
}
